// MockFusedQKVAttention_28638841929834
// MI455X (gfx1250) — matrix operand order repaired and checked by reading, not compiled, not run on hardware
//
#include <hip/hip_runtime.h>
#include <stdint.h>
#include <stddef.h>

// ---------------------------------------------------------------------------
// Problem constants (MockFusedQKVAttention): B=2, S=2048, H=1024, NH=16, HD=64
// ---------------------------------------------------------------------------
#define HSZ   1024
#define NHEAD 16
#define HDIM  64
#define BATCH 2
#define SEQ   2048
#define TOK   (BATCH * SEQ)   // 4096 tokens
#define LDSS  36              // LDS row stride (bf16 elems): 72B rows, 8B aligned, conflict-free

typedef __attribute__((ext_vector_type(16))) __bf16   v16bf;
typedef __attribute__((ext_vector_type(8)))  float    v8f;
typedef __attribute__((ext_vector_type(8)))  uint32_t v8u;

// ---------------------------------------------------------------------------
// CDNA5 async global->LDS copy (ASYNCcnt-tracked), per ISA 15.18.3 opcode 97.
// vdst = per-lane LDS byte offset, vaddr = 64-bit global address (GV mode).
// ---------------------------------------------------------------------------
static __device__ __forceinline__ void async_copy_b64(const __bf16* gsrc, __bf16* ldst) {
  asm volatile("global_load_async_to_lds_b64 %0, %1, off"
               :: "v"((uint32_t)(uintptr_t)(void*)ldst),
                  "v"((uint64_t)(uintptr_t)(const void*)gsrc)
               : "memory");
}
static __device__ __forceinline__ void wait_async0() {
  asm volatile("s_wait_asynccnt 0x0" ::: "memory");
}

// ---------------------------------------------------------------------------
// Fragment loaders (CDNA5 WMMA VGPR layouts, wave32)
// ---------------------------------------------------------------------------
static __device__ __forceinline__ v16bf load_a_frag(const __bf16* base, int stride) {
  const int lane = threadIdx.x & 31;
  const int hf   = lane >> 4;
  const int m    = lane & 15;
  const __bf16* row = base + m * stride + 8 * hf;
  v8u u;
#pragma unroll
  for (int j = 0; j < 4; ++j)
    u[j] = *reinterpret_cast<const uint32_t*>(row + 2 * j);
#pragma unroll
  for (int j = 0; j < 4; ++j)
    u[4 + j] = *reinterpret_cast<const uint32_t*>(row + 16 + 2 * j);
  return __builtin_bit_cast(v16bf, u);
}

// B-fragment (32x16, KxN) read from an [N][K] row-major buffer
static __device__ __forceinline__ v16bf load_b_frag(const __bf16* base, int stride) {
  const int lane = threadIdx.x & 31;
  const int hf   = lane >> 4;
  const int n    = lane & 15;
  const __bf16* row = base + n * stride + 8 * hf;
  v8u u;
#pragma unroll
  for (int j = 0; j < 4; ++j)
    u[j] = *reinterpret_cast<const uint32_t*>(row + 2 * j);
#pragma unroll
  for (int j = 0; j < 4; ++j)
    u[4 + j] = *reinterpret_cast<const uint32_t*>(row + 16 + 2 * j);
  return __builtin_bit_cast(v16bf, u);
}

static __device__ __forceinline__ v8f wmma_bf16(v16bf a, v16bf b, v8f c) {
  return __builtin_amdgcn_wmma_f32_16x16x32_bf16(false, a, false, b, (short)0, c,
                                                 false, false);
}

// ---------------------------------------------------------------------------
// Async-stage one K-step: A tile 256x32 bf16, B tile 128x32 bf16 (both [*][K])
// 256 threads; b64 chunks (4 bf16). LDS rows 72B apart -> every chunk 8B aligned.
// ---------------------------------------------------------------------------
static __device__ __forceinline__ void stage_tiles(const __bf16* __restrict__ Ag,
                                                   const __bf16* __restrict__ Bg,
                                                   __bf16* As, __bf16* Bs,
                                                   int m_blk, int n_blk, int K, int k0) {
  const int tid = threadIdx.x;
#pragma unroll
  for (int i = 0; i < 8; ++i) {           // 2048 chunks = 256 rows x 8 chunks
    int flat = tid + i * 256;
    int row  = flat >> 3;
    int c4   = (flat & 7) * 4;
    async_copy_b64(Ag + (size_t)(m_blk + row) * K + k0 + c4, As + row * LDSS + c4);
  }
#pragma unroll
  for (int i = 0; i < 4; ++i) {           // 1024 chunks = 128 rows x 8 chunks
    int flat = tid + i * 256;
    int row  = flat >> 3;
    int c4   = (flat & 7) * 4;
    async_copy_b64(Bg + (size_t)(n_blk + row) * K + k0 + c4, Bs + row * LDSS + c4);
  }
}

// ---------------------------------------------------------------------------
// Double-buffered WMMA GEMM core: C[256x128] += A[256xK] * Bt[128xK]^T
// 8 waves: wave -> (wm 0..3)*64 rows x (wn 0..1)*64 cols ; acc[4][4]
// ---------------------------------------------------------------------------
static __device__ __forceinline__ void gemm_core(const __bf16* __restrict__ Ag,
                                                 const __bf16* __restrict__ Bg,
                                                 __bf16* As0, __bf16* As1,
                                                 __bf16* Bs0, __bf16* Bs1,
                                                 int m_blk, int n_blk, int K,
                                                 v8f acc[4][4]) {
  const int wave = threadIdx.x >> 5;
  const int wm   = wave >> 1;
  const int wn   = wave & 1;

  stage_tiles(Ag, Bg, As0, Bs0, m_blk, n_blk, K, 0);

  for (int k0 = 0; k0 < K; k0 += 32) {
    const int cur = (k0 >> 5) & 1;
    __bf16* Asc = cur ? As1 : As0;
    __bf16* Bsc = cur ? Bs1 : Bs0;

    wait_async0();        // this wave's async writes (incl. last prefetch) done
    __syncthreads();      // -> every wave's writes visible; prior reads retired

    if (k0 + 32 < K)      // overlap next tile's DMA with this tile's WMMAs
      stage_tiles(Ag, Bg, cur ? As0 : As1, cur ? Bs0 : Bs1,
                  m_blk, n_blk, K, k0 + 32);

    v16bf af[4], bfr[4];
#pragma unroll
    for (int mi = 0; mi < 4; ++mi)
      af[mi] = load_a_frag(Asc + (wm * 64 + mi * 16) * LDSS, LDSS);
#pragma unroll
    for (int ni = 0; ni < 4; ++ni)
      bfr[ni] = load_b_frag(Bsc + (wn * 64 + ni * 16) * LDSS, LDSS);
#pragma unroll
    for (int mi = 0; mi < 4; ++mi)
#pragma unroll
      for (int ni = 0; ni < 4; ++ni)
        acc[mi][ni] = wmma_bf16(af[mi], bfr[ni], acc[mi][ni]);
  }
}

// ---------------------------------------------------------------------------
// Kernel 1a: f32 [K][N] -> bf16 [N][K] transpose/convert (weights, one-time)
// ---------------------------------------------------------------------------
__global__ void __launch_bounds__(256) k_transpose_bf16(const float* __restrict__ in,
                                                        __bf16* __restrict__ out,
                                                        int Kdim, int Ndim) {
  size_t idx = (size_t)blockIdx.x * blockDim.x + threadIdx.x;
  size_t total = (size_t)Kdim * Ndim;
  if (idx >= total) return;
  size_t k = idx / Ndim;
  size_t n = idx % Ndim;
  out[n * (size_t)Kdim + k] = (__bf16)in[idx];
}

// Kernel 1b: straight f32 -> bf16 convert (activations)
__global__ void __launch_bounds__(256) k_convert_bf16(const float* __restrict__ in,
                                                      __bf16* __restrict__ out,
                                                      size_t total) {
  size_t idx = (size_t)blockIdx.x * blockDim.x + threadIdx.x;
  if (idx < total) out[idx] = (__bf16)in[idx];
}

// ---------------------------------------------------------------------------
// Kernel 2: QKV GEMM  Xb[4096x1024]bf16 * W1t^T + b_attn
//   epilogue scatters: Q,K -> bf16 [B*NH][S][64] ; V -> bf16 [B*NH][64][S]
// ---------------------------------------------------------------------------
__global__ void __launch_bounds__(256) k_qkv_gemm(const __bf16* __restrict__ Xb,
                                                  const __bf16* __restrict__ W1t,
                                                  const float* __restrict__ b_attn,
                                                  __bf16* __restrict__ qws,
                                                  __bf16* __restrict__ kws,
                                                  __bf16* __restrict__ vws) {
  __shared__ __bf16 As[2][256 * LDSS];
  __shared__ __bf16 Bs[2][128 * LDSS];
  v8f acc[4][4];
  const v8f z = {0.f, 0.f, 0.f, 0.f, 0.f, 0.f, 0.f, 0.f};
#pragma unroll
  for (int i = 0; i < 4; ++i)
#pragma unroll
    for (int j = 0; j < 4; ++j) acc[i][j] = z;

  const int m_blk = blockIdx.y * 256;
  const int n_blk = blockIdx.x * 128;
  gemm_core(Xb, W1t, As[0], As[1], Bs[0], Bs[1], m_blk, n_blk, HSZ, acc);

  const int wave = threadIdx.x >> 5, wm = wave >> 1, wn = wave & 1;
  const int lane = threadIdx.x & 31, hf = lane >> 4, nl = lane & 15;
#pragma unroll
  for (int mi = 0; mi < 4; ++mi)
#pragma unroll
    for (int ni = 0; ni < 4; ++ni)
#pragma unroll
      for (int r = 0; r < 8; ++r) {
        int m = m_blk + wm * 64 + mi * 16 + r + 8 * hf;   // token
        int c = n_blk + wn * 64 + ni * 16 + nl;           // 0..3071
        float v = acc[mi][ni][r] + b_attn[c];
        int part = c >> 10;            // 0=Q 1=K 2=V
        int hc = c & 1023;
        int h = hc >> 6, d = hc & 63;
        int b = m >> 11, s = m & 2047;
        int bh = b * NHEAD + h;
        __bf16 bv = (__bf16)v;
        if (part == 0)      qws[((size_t)bh * SEQ + s) * HDIM + d] = bv;
        else if (part == 1) kws[((size_t)bh * SEQ + s) * HDIM + d] = bv;
        else                vws[((size_t)bh * HDIM + d) * SEQ + s] = bv;
      }
}

// ---------------------------------------------------------------------------
// Kernel 3: flash attention. grid = (S/64, B*NH), 128 threads = 4 waves.
// Each wave owns 16 query rows; streams keys in chunks of 32 (L2-resident).
// ---------------------------------------------------------------------------
__global__ void __launch_bounds__(128) k_attention(const __bf16* __restrict__ qws,
                                                   const __bf16* __restrict__ kws,
                                                   const __bf16* __restrict__ vws,
                                                   __bf16* __restrict__ attn_out) {
  const int bh   = blockIdx.y;              // 0..31
  const int qt   = blockIdx.x;              // 0..31
  const int wave = threadIdx.x >> 5;        // 0..3
  const int lane = threadIdx.x & 31;
  const int hf   = lane >> 4, nl = lane & 15;
  const int q0   = qt * 64 + wave * 16;

  __shared__ __bf16 Pl[4][16 * 34];         // per-wave 16x32 P staging tile

  const __bf16* qbase = qws + ((size_t)bh * SEQ + q0) * HDIM;
  const __bf16* kbase = kws + (size_t)bh * SEQ * HDIM;
  const __bf16* vbase = vws + (size_t)bh * HDIM * SEQ;

  const v16bf aq0 = load_a_frag(qbase, HDIM);       // Q d0..31
  const v16bf aq1 = load_a_frag(qbase + 32, HDIM);  // Q d32..63

  const v8f z = {0.f, 0.f, 0.f, 0.f, 0.f, 0.f, 0.f, 0.f};
  v8f o[4];
#pragma unroll
  for (int dt = 0; dt < 4; ++dt) o[dt] = z;
  float runm[8], runl[8];
#pragma unroll
  for (int r = 0; r < 8; ++r) { runm[r] = -__builtin_inff(); runl[r] = 0.f; }

  const float scale = 0.125f;  // 1/sqrt(64)

  for (int kc = 0; kc < SEQ; kc += 32) {
    v16bf bk00 = load_b_frag(kbase + (size_t)kc * HDIM, HDIM);
    v16bf bk01 = load_b_frag(kbase + (size_t)kc * HDIM + 32, HDIM);
    v16bf bk10 = load_b_frag(kbase + (size_t)(kc + 16) * HDIM, HDIM);
    v16bf bk11 = load_b_frag(kbase + (size_t)(kc + 16) * HDIM + 32, HDIM);
    v8f s0 = z, s1 = z;
    s0 = wmma_bf16(aq0, bk00, s0);
    s0 = wmma_bf16(aq1, bk01, s0);
    s1 = wmma_bf16(aq0, bk10, s1);
    s1 = wmma_bf16(aq1, bk11, s1);

    // online softmax; C/D rows live in 16-lane halves -> xor 1/2/4/8 reduction
    float p0[8], p1[8], corr[8];
#pragma unroll
    for (int r = 0; r < 8; ++r) {
      float t0 = s0[r] * scale;
      float t1 = s1[r] * scale;
      float mx = fmaxf(t0, t1);
      mx = fmaxf(mx, __shfl_xor(mx, 1));
      mx = fmaxf(mx, __shfl_xor(mx, 2));
      mx = fmaxf(mx, __shfl_xor(mx, 4));
      mx = fmaxf(mx, __shfl_xor(mx, 8));
      float nm = fmaxf(runm[r], mx);
      corr[r] = __expf(runm[r] - nm);
      runm[r] = nm;
      p0[r] = __expf(t0 - nm);
      p1[r] = __expf(t1 - nm);
      float rs = p0[r] + p1[r];
      rs += __shfl_xor(rs, 1);
      rs += __shfl_xor(rs, 2);
      rs += __shfl_xor(rs, 4);
      rs += __shfl_xor(rs, 8);
      runl[r] = runl[r] * corr[r] + rs;
    }
#pragma unroll
    for (int dt = 0; dt < 4; ++dt)
#pragma unroll
      for (int r = 0; r < 8; ++r) o[dt][r] *= corr[r];

    // re-layout P (C-layout) -> A-fragment via per-wave LDS tile (in-order DS)
    __bf16* pw = &Pl[wave][0];
#pragma unroll
    for (int r = 0; r < 8; ++r) {
      int row = r + 8 * hf;
      pw[row * 34 + nl]      = (__bf16)p0[r];
      pw[row * 34 + 16 + nl] = (__bf16)p1[r];
    }
    v16bf ap = load_a_frag(pw, 34);

    // O += P * V   (V transposed [d][S] -> contiguous B-fragments)
#pragma unroll
    for (int dt = 0; dt < 4; ++dt) {
      v16bf bv = load_b_frag(vbase + (size_t)(dt * 16) * SEQ + kc, SEQ);
      o[dt] = wmma_bf16(ap, bv, o[dt]);
    }
  }

  const int b = bh >> 4, h = bh & 15;
#pragma unroll
  for (int dt = 0; dt < 4; ++dt)
#pragma unroll
    for (int r = 0; r < 8; ++r) {
      int srow = q0 + r + 8 * hf;
      size_t t = (size_t)b * SEQ + srow;
      attn_out[t * HSZ + h * HDIM + dt * 16 + nl] = (__bf16)(o[dt][r] / runl[r]);
    }
}

// ---------------------------------------------------------------------------
// Kernel 4: projection GEMM  attn[4096x1024]bf16 * W2t^T + b_proj -> f32 out
// ---------------------------------------------------------------------------
__global__ void __launch_bounds__(256) k_proj_gemm(const __bf16* __restrict__ A,
                                                   const __bf16* __restrict__ W2t,
                                                   const float* __restrict__ b_proj,
                                                   float* __restrict__ out) {
  __shared__ __bf16 As[2][256 * LDSS];
  __shared__ __bf16 Bs[2][128 * LDSS];
  v8f acc[4][4];
  const v8f z = {0.f, 0.f, 0.f, 0.f, 0.f, 0.f, 0.f, 0.f};
#pragma unroll
  for (int i = 0; i < 4; ++i)
#pragma unroll
    for (int j = 0; j < 4; ++j) acc[i][j] = z;

  const int m_blk = blockIdx.y * 256;
  const int n_blk = blockIdx.x * 128;
  gemm_core(A, W2t, As[0], As[1], Bs[0], Bs[1], m_blk, n_blk, HSZ, acc);

  const int wave = threadIdx.x >> 5, wm = wave >> 1, wn = wave & 1;
  const int lane = threadIdx.x & 31, hf = lane >> 4, nl = lane & 15;
#pragma unroll
  for (int mi = 0; mi < 4; ++mi)
#pragma unroll
    for (int ni = 0; ni < 4; ++ni)
#pragma unroll
      for (int r = 0; r < 8; ++r) {
        int m = m_blk + wm * 64 + mi * 16 + r + 8 * hf;
        int c = n_blk + wn * 64 + ni * 16 + nl;
        out[(size_t)m * HSZ + c] = acc[mi][ni][r] + b_proj[c];
      }
}

// ---------------------------------------------------------------------------
// Launcher
// ---------------------------------------------------------------------------
extern "C" void kernel_launch(void* const* d_in, const int* in_sizes, int n_in,
                              void* d_out, int out_size, void* d_ws, size_t ws_size,
                              hipStream_t stream) {
  (void)in_sizes; (void)n_in; (void)out_size; (void)ws_size;

  const float* hidden = (const float*)d_in[0];  // [2,2048,1024]
  const float* w_attn = (const float*)d_in[1];  // [1024,3072]
  const float* b_attn = (const float*)d_in[2];  // [3072]
  const float* w_proj = (const float*)d_in[3];  // [1024,1024]
  const float* b_proj = (const float*)d_in[4];  // [1024]
  float* out = (float*)d_out;                    // [2,2048,1024]

  // Workspace layout (bf16 elements), ~50 MB total
  const size_t HW  = (size_t)HSZ * HSZ;          // 1M
  const size_t PHD = (size_t)BATCH * NHEAD * SEQ * HDIM;  // 4M (per q/k/v array)
  __bf16* W1t  = (__bf16*)d_ws;        // [3072][1024]
  __bf16* W2t  = W1t + 3 * HW;         // [1024][1024]
  __bf16* Xb   = W2t + HW;             // [4096][1024]
  __bf16* qws  = Xb + (size_t)TOK * HSZ;
  __bf16* kws  = qws + PHD;
  __bf16* vws  = kws + PHD;
  __bf16* attn = vws + PHD;            // [4096][1024]

  // 1) one-time converts/transposes
  {
    size_t tot1 = 3 * HW;
    k_transpose_bf16<<<dim3((unsigned)((tot1 + 255) / 256)), dim3(256), 0, stream>>>(
        w_attn, W1t, HSZ, 3 * HSZ);
    k_transpose_bf16<<<dim3((unsigned)((HW + 255) / 256)), dim3(256), 0, stream>>>(
        w_proj, W2t, HSZ, HSZ);
    size_t totx = (size_t)TOK * HSZ;
    k_convert_bf16<<<dim3((unsigned)((totx + 255) / 256)), dim3(256), 0, stream>>>(
        hidden, Xb, totx);
  }
  // 2) QKV GEMM + head scatter
  k_qkv_gemm<<<dim3(3 * HSZ / 128, TOK / 256), dim3(256), 0, stream>>>(
      Xb, W1t, b_attn, qws, kws, vws);
  // 3) attention
  k_attention<<<dim3(SEQ / 64, BATCH * NHEAD), dim3(128), 0, stream>>>(
      qws, kws, vws, attn);
  // 4) projection
  k_proj_gemm<<<dim3(HSZ / 128, TOK / 256), dim3(256), 0, stream>>>(
      attn, W2t, b_proj, out);
}
